// BackpropWiSARD_67276367725205
// MI455X (gfx1250) — compile-verified, matrix operand order checked by reading, not run on hardware
//
#include <hip/hip_runtime.h>
#include <stdint.h>

typedef int v8i __attribute__((ext_vector_type(8)));

// Problem dims (fixed by the reference)
constexpr int B = 512, C = 100, F = 224, E = 8192, H = 4, I = 28;
constexpr int NBITS = 6272;                 // x_b columns == F*I (no padding needed)
constexpr int WPR = E / 32;                 // 256 words of sign bits per (c,f) row
constexpr long long NWORDS = (long long)C * F * WPR;  // 5,734,400 words = 22.9 MB

// ---------------------------------------------------------------------------
// Kernel A: pack sign bits of table (bit=1 iff table[e] >= 0).
// One streaming pass over 734 MB; wave32 ballot packs 32 lanes -> 1 word.
// ---------------------------------------------------------------------------
__global__ void pack_signs(const float* __restrict__ table, uint32_t* __restrict__ bits) {
  size_t g = (size_t)blockIdx.x * blockDim.x + threadIdx.x;
  float v = table[g];
  uint32_t m = __builtin_amdgcn_ballot_w32(v >= 0.0f);
  if ((threadIdx.x & 31u) == 0u) bits[g >> 5] = m;
}

// ---------------------------------------------------------------------------
// Kernel B: H3 hash. One thread per (b,f): XOR-fold hash_values over the 28
// permuted input bits; emit 4 indices in [0, 8192).
// ---------------------------------------------------------------------------
__global__ void h3_hash_kernel(const int* __restrict__ xb,
                               const int* __restrict__ order,
                               const int* __restrict__ hv,
                               int* __restrict__ idxbuf) {
  __shared__ int hvs[H * I];
  if (threadIdx.x < H * I) hvs[threadIdx.x] = hv[threadIdx.x];
  __syncthreads();
  int g = blockIdx.x * blockDim.x + threadIdx.x;   // g = b*F + f
  int b = g / F, f = g - b * F;
  const int* ob   = order + f * I;
  const int* xrow = xb + (size_t)b * NBITS;
  int a0 = 0, a1 = 0, a2 = 0, a3 = 0;
#pragma unroll
  for (int i = 0; i < I; ++i) {
    int col = ob[i];
    int msk = (xrow[col] != 0) ? -1 : 0;           // branchless XOR-accumulate
    a0 ^= hvs[0 * I + i] & msk;
    a1 ^= hvs[1 * I + i] & msk;
    a2 ^= hvs[2 * I + i] & msk;
    a3 ^= hvs[3 * I + i] & msk;
  }
  ((int4*)idxbuf)[g] = make_int4(a0, a1, a2, a3);
}

// ---------------------------------------------------------------------------
// Kernel C: main. One block per batch b (224 threads = 7 waves).
// Wave w covers classes [16w, 16w+16). Per wave: build int8 sign matrix
// A (16 classes x 64 features) and reduce over features with
// v_wmma_i32_16x16x64_iu8 against an all-ones B matrix (4 chunks cover 224
// features padded to 256). Hash indices staged via async-to-LDS.
// All bit-table gathers use flat 32-bit word offsets off the uniform base so
// the backend can use saddr+voffset addressing (no 64-bit per-lane adds).
// ---------------------------------------------------------------------------
__global__ void __launch_bounds__(224) wisard_main(const uint32_t* __restrict__ bits,
                                                   const int* __restrict__ idxbuf,
                                                   const float* __restrict__ bias,
                                                   float* __restrict__ out) {
  __shared__ __align__(16) unsigned idx_s[F * H];  // 896 ints = 3.5 KB
  const int b = blockIdx.x;
  const int t = threadIdx.x;

  // Stage this batch's 224x4 hash indices into LDS with the gfx1250 async path.
  {
    unsigned lds_off = (unsigned)(uintptr_t)(&idx_s[t * 4]);
    unsigned long long src = (unsigned long long)(idxbuf + ((size_t)b * F + t) * 4);
    asm volatile("global_load_async_to_lds_b128 %0, %1, off"
                 :: "v"(lds_off), "v"(src) : "memory");
  }
  asm volatile("s_wait_asynccnt 0" ::: "memory");
  __syncthreads();

  const int lane = t & 31;
  const int c0   = (t >> 5) * 16;     // class tile base for this wave
  const int m    = lane & 15;         // WMMA row = class within tile
  const int hi   = lane >> 4;         // lane half selects 32-feature sub-chunk
  const int c    = c0 + m;
  const int cc   = (c < C) ? c : (C - 1);   // clamp for safe gathers; write guarded

  // Word offset of this class's sign-bit block (uniform SGPR base + 32-bit lane offset)
  const uint32_t rowc = ((uint32_t)cc * (uint32_t)F) << 8;   // * 256 words per row

  v8i acc = {0, 0, 0, 0, 0, 0, 0, 0};
  v8i ones;
#pragma unroll
  for (int k = 0; k < 8; ++k) ones[k] = 0x01010101;  // B matrix: all 1s (u8)

#pragma unroll
  for (int chunk = 0; chunk < 4; ++chunk) {
    const int fbase = chunk * 64 + hi * 32;
    int aw[8];
#pragma unroll
    for (int k = 0; k < 8; ++k) aw[k] = 0;
#pragma unroll
    for (int j = 0; j < 32; ++j) {
      int f          = fbase + j;
      int valid      = (f < F);
      uint32_t fc    = valid ? (uint32_t)f : 0u;
      uint4 iv = *(const uint4*)(&idx_s[fc << 2]);          // 4 hash indices (>=0)
      uint32_t rb = rowc + (fc << 8);                       // 32-bit word offset
      uint32_t b0 = bits[rb + (iv.x >> 5)] >> (iv.x & 31u);
      uint32_t b1 = bits[rb + (iv.y >> 5)] >> (iv.y & 31u);
      uint32_t b2 = bits[rb + (iv.z >> 5)] >> (iv.z & 31u);
      uint32_t b3 = bits[rb + (iv.w >> 5)] >> (iv.w & 31u);
      int pos = (int)(b0 & b1 & b2 & b3 & 1u);              // all 4 entries >= 0 ?
      int sb  = valid ? (pos ? 1 : -1) : 0;                 // int8 sign, 0-pad tail
      aw[j >> 2] |= (sb & 0xFF) << ((j & 3) * 8);
    }
    v8i av;
#pragma unroll
    for (int k = 0; k < 8; ++k) av[k] = aw[k];
    // D = A(i8 signed) x ones(u8) + C : row sums == sum_f s[b,c,f]
    acc = __builtin_amdgcn_wmma_i32_16x16x64_iu8(true, av, false, ones, acc,
                                                 false, false);
  }

  // C/D layout: VGPR r holds M=r (lanes 0-15) / M=r+8 (lanes 16-31); all N equal.
  if ((lane & 15) == 0) {
    const int rbase = hi * 8;
#pragma unroll
    for (int r = 0; r < 8; ++r) {
      int cr = c0 + rbase + r;
      if (cr < C) out[(size_t)b * C + cr] = (float)acc[r] + bias[cr];
    }
  }
}

// ---------------------------------------------------------------------------
extern "C" void kernel_launch(void* const* d_in, const int* in_sizes, int n_in,
                              void* d_out, int out_size, void* d_ws, size_t ws_size,
                              hipStream_t stream) {
  const int*   xb    = (const int*)d_in[0];     // (512, 6272) int32
  const int*   order = (const int*)d_in[1];     // (6272,) int32 permutation
  const int*   hv    = (const int*)d_in[2];     // (4, 28) int32
  const float* table = (const float*)d_in[3];   // (100, 224, 8192) f32
  const float* bias  = (const float*)d_in[4];   // (100,) f32
  float* out = (float*)d_out;                   // (512, 100) f32

  uint32_t* bits   = (uint32_t*)d_ws;                            // 22,937,600 B
  int*      idxbuf = (int*)((char*)d_ws + (size_t)NWORDS * 4);   // + 1,835,008 B

  // A: 183.5M elements / 256 threads = 716,800 blocks (exact)
  pack_signs<<<(int)(NWORDS * 32 / 256), 256, 0, stream>>>(table, bits);
  // B: 512*224 = 114,688 threads = 448 blocks (exact)
  h3_hash_kernel<<<(B * F) / 256, 256, 0, stream>>>(xb, order, hv, idxbuf);
  // C: one block per batch, 7 waves of 16 classes each
  wisard_main<<<B, 224, 0, stream>>>(bits, idxbuf, bias, out);
}
